// LeapfrogIntegrator_72593537237648
// MI455X (gfx1250) — matrix operand order repaired and checked by reading, not compiled, not run on hardware
//
#include <hip/hip_runtime.h>

typedef __bf16 bf16_t;
typedef __attribute__((ext_vector_type(16))) __bf16 v16bf;
typedef __attribute__((ext_vector_type(8)))  __bf16 v8bf;
typedef __attribute__((ext_vector_type(8)))  float  v8f;

union ABFrag { v16bf v; v8bf h[2]; };

#define DT_CONST 0.01f

// Problem geometry (fixed by the reference)
#define B_ROWS 4096
#define D_DIM  1024
#define R_DIM  256
#define MTILE  16
#define NT1    (R_DIM / 16)   // 16 n-tiles in GEMM1 (u = v @ U^T)
#define NT2    (D_DIM / 16)   // 64 n-tiles in GEMM2 (Gamma = s @ W^T)
#define KC1    (D_DIM / 32)   // 32 K-chunks of 32 in GEMM1
#define KC2    (R_DIM / 32)   //  8 K-chunks of 32 in GEMM2
#define WAVES  16
#define TPB    (WAVES * 32)

// LDS row strides padded to spread rows across banks (row stride in dwords % 64 == 4)
#define VBS_STRIDE (D_DIM + 8)   // bf16 copy of v tile
#define SS_STRIDE  (R_DIM + 8)   // bf16 copy of u^2 tile

#define UPK_ELEMS (NT1 * KC1 * 32 * 16)   // 262144 bf16 = 512 KB
#define WPK_ELEMS (NT2 * KC2 * 32 * 16)   // 262144 bf16 = 512 KB
#define CSTRIDE   512                     // bf16 elements per (chunk) per n-tile stream

// ---------------------------------------------------------------------------
// Pack U (R x D, f32) and W (D x R, f32) into bf16 WMMA B-fragment order:
//   frag(t, c): lane l holds, in VGPR j, pair B[32c + 16*(l>>4) + 2j + {0,1}, 16t + (l&15)]
// Flat index f = ((t*KC + c)*32 + l)*16 + e  with  k_local = 16*(l>>4) + e.
// GEMM1: B[k,n] = U[n,k];   GEMM2: B[k,n] = W[n,k].
// ---------------------------------------------------------------------------
__global__ void pack_uw_kernel(const float* __restrict__ U, const float* __restrict__ W,
                               bf16_t* __restrict__ Upk, bf16_t* __restrict__ Wpk) {
  const int total = UPK_ELEMS + WPK_ELEMS;
  for (int f = blockIdx.x * blockDim.x + threadIdx.x; f < total;
       f += gridDim.x * blockDim.x) {
    if (f < UPK_ELEMS) {
      const int e = f & 15;
      const int l = (f >> 4) & 31;
      const int c = (f >> 9) & (KC1 - 1);
      const int t = f >> 14;
      const int n = 16 * t + (l & 15);              // r index in U
      const int k = 32 * c + 16 * (l >> 4) + e;     // d index in U
      Upk[f] = (bf16_t)U[n * D_DIM + k];
    } else {
      const int g = f - UPK_ELEMS;
      const int e = g & 15;
      const int l = (g >> 4) & 31;
      const int c = (g >> 9) & (KC2 - 1);
      const int t = g >> 12;
      const int n = 16 * t + (l & 15);              // d index in W
      const int k = 32 * c + 16 * (l >> 4) + e;     // r index in W
      Wpk[g] = (bf16_t)W[n * R_DIM + k];
    }
  }
}

// A-fragment from an LDS row (16 contiguous bf16: k = 16*kh + e, matching packed B)
__device__ __forceinline__ void load_a_frag(ABFrag& a, const bf16_t* row, int c, int kh) {
  a.h[0] = *(const v8bf*)(row + 32 * c + 16 * kh);
  a.h[1] = *(const v8bf*)(row + 32 * c + 16 * kh + 8);
}

// B-fragment: 32 contiguous bytes per lane; uniform base + 32-bit element offset
__device__ __forceinline__ void load_b_frag(ABFrag& b, const bf16_t* __restrict__ base,
                                            int off) {
  b.h[0] = *(const v8bf*)(base + off);
  b.h[1] = *(const v8bf*)(base + off + 8);
}

__device__ __forceinline__ v8f wmma_bf16(const ABFrag& a, const ABFrag& b, v8f c) {
  return __builtin_amdgcn_wmma_f32_16x16x32_bf16(false, a.v, false, b.v,
                                                 (short)0, c, false, false);
}

// ---------------------------------------------------------------------------
// Persistent fused leapfrog: each workgroup owns a 16-row tile for all steps.
// ---------------------------------------------------------------------------
__global__ __launch_bounds__(TPB)
void leapfrog_fused_kernel(const float* __restrict__ x_in,
                           const float* __restrict__ v_in,
                           const float* __restrict__ f_in,
                           const bf16_t* __restrict__ Upk,
                           const bf16_t* __restrict__ Wpk,
                           const int* __restrict__ steps_ptr,
                           float* __restrict__ out) {
  __shared__ bf16_t vb_s[MTILE * VBS_STRIDE];   // bf16 copy of current v tile (WMMA A source)
  __shared__ bf16_t s_s [MTILE * SS_STRIDE];    // bf16 u^2 tile (WMMA A source for GEMM2)

  const int tid  = threadIdx.x;
  const int wave = tid >> 5;
  const int lane = tid & 31;
  const int nl   = lane & 15;    // N index within 16x16 C tile / B fragment
  const int kh   = lane >> 4;    // half-wave selector (K-group / M-group)
  const int rowbase = blockIdx.x * MTILE;

  const float dt  = DT_CONST;          // DT * DT_SCALE
  const float hdt = 0.5f * DT_CONST;
  const int nsteps = steps_ptr[0];

  // Register masters for the 32 elements this lane owns:
  // element (tt, i): m = i + 8*kh, n = 16*(wave*4 + tt) + nl
  float vreg[32], freg[32], xreg[32];

  #pragma unroll
  for (int tt = 0; tt < 4; ++tt) {
    const int t2 = wave * 4 + tt;
    #pragma unroll
    for (int i = 0; i < 8; ++i) {
      const int m = i + 8 * kh;
      const int n = 16 * t2 + nl;
      const int r = tt * 8 + i;
      const size_t g = (size_t)(rowbase + m) * D_DIM + n;
      const float vv = v_in[g];
      vreg[r] = vv;
      freg[r] = f_in[g];
      xreg[r] = x_in[g];
      vb_s[m * VBS_STRIDE + n] = (bf16_t)vv;
    }
  }
  __syncthreads();

  const int ma = nl;  // A-fragment row handled by this lane (m = lane & 15)
  const bf16_t* arow = &vb_s[ma * VBS_STRIDE];
  const bf16_t* srow = &s_s [ma * SS_STRIDE];
  const int ub0 = (wave * KC1 * 32 + lane) * 16;        // GEMM1 B stream base (elements)
  const int wb0 = ((wave * 4) * KC2 * 32 + lane) * 16;  // GEMM2 B stream base (elements)

  for (int s = 0; s < nsteps; ++s) {
    #pragma unroll
    for (int half = 0; half < 2; ++half) {
      // ---------------- GEMM1: u[:, 16w:16w+16] = v_tile @ U^T ---------------
      // Strict 1-deep double buffer, dual accumulators, hardware loop so at
      // most two fragment sets are ever in flight (no spill-inducing hoists).
      v8f accA, accB;
      #pragma unroll
      for (int i = 0; i < 8; ++i) { accA[i] = 0.0f; accB[i] = 0.0f; }

      ABFrag a0, a1, b0, b1;
      load_a_frag(a0, arow, 0, kh);
      load_b_frag(b0, Upk, ub0);
      #pragma unroll 1
      for (int c = 0; c < KC1 - 2; c += 2) {
        __builtin_prefetch((const void*)(Upk + ub0 + (c + 8) * CSTRIDE), 0, 3);
        load_a_frag(a1, arow, c + 1, kh);
        load_b_frag(b1, Upk, ub0 + (c + 1) * CSTRIDE);
        accA = wmma_bf16(a0, b0, accA);          // chunk c
        load_a_frag(a0, arow, c + 2, kh);
        load_b_frag(b0, Upk, ub0 + (c + 2) * CSTRIDE);
        accB = wmma_bf16(a1, b1, accB);          // chunk c+1
      }
      // epilogue: a0/b0 hold chunk KC1-2
      load_a_frag(a1, arow, KC1 - 1, kh);
      load_b_frag(b1, Upk, ub0 + (KC1 - 1) * CSTRIDE);
      accA = wmma_bf16(a0, b0, accA);
      accB = wmma_bf16(a1, b1, accB);
      const v8f acc = accA + accB;

      // s = u*u  -> LDS in bf16 (C layout: m = i + 8*kh, n = 16*wave + nl)
      #pragma unroll
      for (int i = 0; i < 8; ++i) {
        const float u = acc[i];
        s_s[(i + 8 * kh) * SS_STRIDE + 16 * wave + nl] = (bf16_t)(u * u);
      }
      __syncthreads();

      // ---------------- GEMM2: Gamma[:, 16t2:16t2+16] = s @ W^T --------------
      // Two tile-groups of 2 to cap register pressure; each group does its
      // 2 accumulator chains with 1-deep pipelined fragments, then updates
      // its owned elements immediately (writes only vb_s / registers).
      #pragma unroll
      for (int grp = 0; grp < 2; ++grp) {
        v8f acc2a, acc2b;
        #pragma unroll
        for (int i = 0; i < 8; ++i) { acc2a[i] = 0.0f; acc2b[i] = 0.0f; }

        const int wba = wb0 + (grp * 2)     * KC2 * CSTRIDE;  // tile t2 = wave*4+2g
        const int wbb = wb0 + (grp * 2 + 1) * KC2 * CSTRIDE;  // tile t2+1

        ABFrag sa0, sa1, ba0, bb0, ba1, bb1;
        load_a_frag(sa0, srow, 0, kh);
        load_b_frag(ba0, Wpk, wba);
        load_b_frag(bb0, Wpk, wbb);
        #pragma unroll 1
        for (int c = 0; c < KC2 - 2; c += 2) {
          load_a_frag(sa1, srow, c + 1, kh);
          load_b_frag(ba1, Wpk, wba + (c + 1) * CSTRIDE);
          load_b_frag(bb1, Wpk, wbb + (c + 1) * CSTRIDE);
          acc2a = wmma_bf16(sa0, ba0, acc2a);    // chunk c, tile A
          acc2b = wmma_bf16(sa0, bb0, acc2b);    // chunk c, tile B
          load_a_frag(sa0, srow, c + 2, kh);
          load_b_frag(ba0, Wpk, wba + (c + 2) * CSTRIDE);
          load_b_frag(bb0, Wpk, wbb + (c + 2) * CSTRIDE);
          acc2a = wmma_bf16(sa1, ba1, acc2a);    // chunk c+1, tile A
          acc2b = wmma_bf16(sa1, bb1, acc2b);    // chunk c+1, tile B
        }
        // epilogue: sa0/ba0/bb0 hold chunk KC2-2
        load_a_frag(sa1, srow, KC2 - 1, kh);
        load_b_frag(ba1, Wpk, wba + (KC2 - 1) * CSTRIDE);
        load_b_frag(bb1, Wpk, wbb + (KC2 - 1) * CSTRIDE);
        acc2a = wmma_bf16(sa0, ba0, acc2a);
        acc2b = wmma_bf16(sa0, bb0, acc2b);
        acc2a = wmma_bf16(sa1, ba1, acc2a);
        acc2b = wmma_bf16(sa1, bb1, acc2b);

        // -------- leapfrog elementwise update for this group's 2 tiles ------
        // half 0: vh = v + hdt*(F - Gamma); x += dt*vh; v = vh
        // half 1: v  = v + hdt*(F - Gamma)
        #pragma unroll
        for (int q = 0; q < 2; ++q) {
          const v8f& gacc = q ? acc2b : acc2a;
          const int tt = grp * 2 + q;
          #pragma unroll
          for (int i = 0; i < 8; ++i) {
            const int m = i + 8 * kh;
            const int n = 16 * (wave * 4 + tt) + nl;
            const int r = tt * 8 + i;
            const float vh = vreg[r] + hdt * (freg[r] - gacc[i]);
            if (half == 0) xreg[r] += dt * vh;
            vreg[r] = vh;
            vb_s[m * VBS_STRIDE + n] = (bf16_t)vh;
          }
        }
      }
      __syncthreads();
    }
  }

  // ---------------- write (xf, vf) concatenated ------------------------------
  #pragma unroll
  for (int tt = 0; tt < 4; ++tt) {
    #pragma unroll
    for (int i = 0; i < 8; ++i) {
      const int m = i + 8 * kh;
      const int n = 16 * (wave * 4 + tt) + nl;
      const int r = tt * 8 + i;
      const size_t g = (size_t)(rowbase + m) * D_DIM + n;
      out[g] = xreg[r];
      out[(size_t)B_ROWS * D_DIM + g] = vreg[r];
    }
  }
}

// ---------------------------------------------------------------------------
extern "C" void kernel_launch(void* const* d_in, const int* in_sizes, int n_in,
                              void* d_out, int out_size, void* d_ws, size_t ws_size,
                              hipStream_t stream) {
  const float* x     = (const float*)d_in[0];
  const float* v     = (const float*)d_in[1];
  const float* force = (const float*)d_in[2];
  const float* U     = (const float*)d_in[3];
  const float* W     = (const float*)d_in[4];
  const int*   steps = (const int*)d_in[5];

  bf16_t* Upk = (bf16_t*)d_ws;
  bf16_t* Wpk = Upk + UPK_ELEMS;   // needs 1 MB of d_ws total

  pack_uw_kernel<<<512, 256, 0, stream>>>(U, W, Upk, Wpk);

  leapfrog_fused_kernel<<<B_ROWS / MTILE, TPB, 0, stream>>>(
      x, v, force, Upk, Wpk, steps, (float*)d_out);
}
